// SelfAttention_67113158967840
// MI455X (gfx1250) — matrix operand order repaired and checked by reading, not compiled, not run on hardware
//
#include <hip/hip_runtime.h>
#include <hip/hip_bf16.h>
#include <cstdint>
#include <cstddef>

// ---------------------------------------------------------------------------
// Self-attention for MI455X (gfx1250), wave32 + WMMA f16 (f32 accumulate),
// async global->LDS staging for GEMM tiles (GLOBAL_LOAD_ASYNC_TO_LDS_B128).
// B=4, L=2048, C=1024, H=16, D=64.
// ---------------------------------------------------------------------------

typedef __attribute__((ext_vector_type(16))) _Float16 v16h;
typedef __attribute__((ext_vector_type(8)))  float    v8f;
typedef __attribute__((ext_vector_type(4)))  int      v4i;

static constexpr int BB = 4;
static constexpr int LL = 2048;
static constexpr int CC = 1024;
static constexpr int HH = 16;
static constexpr int DD = 64;

#define GLOBAL_AS __attribute__((address_space(1)))
#define LDS_AS    __attribute__((address_space(3)))

typedef GLOBAL_AS v4i* gptr_v4i;
typedef LDS_AS    v4i* lptr_v4i;

#if defined(__has_builtin)
#  if __has_builtin(__builtin_amdgcn_global_load_async_to_lds_b128)
#    define USE_ASYNC_LDS 1
#  endif
#endif
#ifndef USE_ASYNC_LDS
#  define USE_ASYNC_LDS 0
#endif

#if USE_ASYNC_LDS
__device__ __forceinline__ void async_wait0() {
#if __has_builtin(__builtin_amdgcn_s_wait_asynccnt)
  __builtin_amdgcn_s_wait_asynccnt(0);
#else
  asm volatile("s_wait_asynccnt 0x0" ::: "memory");
#endif
}
#endif

__device__ __forceinline__ v8f wmma16(v16h a, v16h b, v8f c) {
  // D = A(16x32 f16) * B(32x16 f16) + C(16x16 f32)
  return __builtin_amdgcn_wmma_f32_16x16x32_f16(
      /*neg_a=*/false, a, /*neg_b=*/false, b,
      /*c_mod=*/(short)0, c, /*reuse_a=*/false, /*reuse_b=*/false);
}

union HFrag {
  v16h  v;
  uint4 u[2];
};

// ---------------------------------------------------------------------------
// fp32 -> f16 elementwise convert (n must be a multiple of 1024).
// ---------------------------------------------------------------------------
__global__ __launch_bounds__(256)
void cvt_f16(const float* __restrict__ src, _Float16* __restrict__ dst)
{
  const int i = (blockIdx.x * blockDim.x + threadIdx.x) * 4;
  const float4 f = *(const float4*)&src[i];
  dst[i + 0] = (_Float16)f.x;
  dst[i + 1] = (_Float16)f.y;
  dst[i + 2] = (_Float16)f.z;
  dst[i + 3] = (_Float16)f.w;
}

// ---------------------------------------------------------------------------
// W[K][N] fp32 -> Wt[N][K] f16 (32x32 tiles through LDS).
// ---------------------------------------------------------------------------
__global__ __launch_bounds__(256)
void transpose_to_f16(const float* __restrict__ W, _Float16* __restrict__ Wt,
                      int K, int N)
{
  __shared__ float tile[32][33];
  const int k0 = blockIdx.y * 32, n0 = blockIdx.x * 32;
  const int tx = threadIdx.x & 31, ty = threadIdx.x >> 5;   // 8 rows/pass
#pragma unroll
  for (int r = ty; r < 32; r += 8)
    tile[r][tx] = W[(size_t)(k0 + r) * N + n0 + tx];
  __syncthreads();
#pragma unroll
  for (int r = ty; r < 32; r += 8)
    Wt[(size_t)(n0 + r) * K + k0 + tx] = (_Float16)tile[tx][r];
}

// ---------------------------------------------------------------------------
// Tiled GEMM:  C[M,N] = A[M,K] * Bt[N,K]^T (+bias), A/Bt f16, C fp32.
// Workgroup tile 128x128, K-step 32; 8 waves in a 2(M) x 4(N) grid; each wave
// owns a 64x32 block = 8 WMMA accumulators.  Both LDS tiles are K-contiguous
// 64B rows, staged with per-lane GLOBAL_LOAD_ASYNC_TO_LDS_B128 (ASYNCcnt)
// when available.
// ---------------------------------------------------------------------------
__global__ __launch_bounds__(256)
void gemm_wmma_f16(const _Float16* __restrict__ A,
                   const _Float16* __restrict__ Bt,
                   float* __restrict__ Cout,
                   const float* __restrict__ bias,
                   int M, int N, int K)
{
  constexpr int TM = 128, TN = 128, TK = 32, PAD = 8;
  __shared__ __align__(16) _Float16 As[TM][TK + PAD];   // [m][k]
  __shared__ __align__(16) _Float16 Bs[TN][TK + PAD];   // [n][k]

  const int tid   = threadIdx.x;
  const int lane  = tid & 31;
  const int wave  = tid >> 5;
  const int half  = lane >> 4;   // K-half selector per ISA A/B layouts
  const int l16   = lane & 15;
  const int waveM = (wave >> 2) * 64;   // 0 / 64
  const int waveN = (wave & 3) * 32;    // 0 / 32 / 64 / 96
  const int m0 = blockIdx.y * TM;
  const int n0 = blockIdx.x * TN;

  v8f acc[4][2];
#pragma unroll
  for (int i = 0; i < 4; ++i)
#pragma unroll
    for (int j = 0; j < 2; ++j) acc[i][j] = v8f{};

  for (int k0 = 0; k0 < K; k0 += TK) {
    // ---- stage A (128x32) and B^T (128x32) tiles: 512 16B chunks each ----
#if USE_ASYNC_LDS
#pragma unroll
    for (int t = 0; t < 2; ++t) {
      const int ch = tid * 2 + t;          // 0..511
      const int r  = ch >> 2;              // 0..127
      const int c  = (ch & 3) * 8;         // f16 elems: 0/8/16/24
      __builtin_amdgcn_global_load_async_to_lds_b128(
          (gptr_v4i)(const void*)&A[(size_t)(m0 + r) * K + k0 + c],
          (lptr_v4i)(void*)&As[r][c], 0, 0);
      __builtin_amdgcn_global_load_async_to_lds_b128(
          (gptr_v4i)(const void*)&Bt[(size_t)(n0 + r) * K + k0 + c],
          (lptr_v4i)(void*)&Bs[r][c], 0, 0);
    }
    async_wait0();
#else
    {
      const int r = tid >> 1;              // 0..127
      const int c = (tid & 1) * 16;        // 0 / 16
      const uint4* sa = (const uint4*)&A[(size_t)(m0 + r) * K + k0 + c];
      *(uint4*)&As[r][c]     = sa[0];
      *(uint4*)&As[r][c + 8] = sa[1];
      const uint4* sb = (const uint4*)&Bt[(size_t)(n0 + r) * K + k0 + c];
      *(uint4*)&Bs[r][c]     = sb[0];
      *(uint4*)&Bs[r][c + 8] = sb[1];
    }
#endif
    if (k0 + TK < K) {  // speculative prefetch of the next K tiles
      __builtin_prefetch(&A[(size_t)(m0 + (tid >> 1)) * K + k0 + TK], 0, 3);
      __builtin_prefetch(&Bt[(size_t)(n0 + (tid >> 1)) * K + k0 + TK], 0, 3);
    }
    __syncthreads();

    // ---- fragments per ISA VGPR layouts ----
    // A 16x32 f16: lane holds row M=l16, K = half*8 + {0..7} and {16..23}.
    HFrag afr[4];
#pragma unroll
    for (int i = 0; i < 4; ++i) {
      const int row = waveM + i * 16 + l16;
      const int kb  = half * 8;
      afr[i].u[0] = *(const uint4*)&As[row][kb];
      afr[i].u[1] = *(const uint4*)&As[row][kb + 16];
    }
    // B 32x16 f16: lane holds col N=l16, K = half*8 + {0..7} and {16..23}.
    HFrag bfr[2];
#pragma unroll
    for (int j = 0; j < 2; ++j) {
      const int col = waveN + j * 16 + l16;
      const int kb  = half * 8;
      bfr[j].u[0] = *(const uint4*)&Bs[col][kb];
      bfr[j].u[1] = *(const uint4*)&Bs[col][kb + 16];
    }
#pragma unroll
    for (int i = 0; i < 4; ++i)
#pragma unroll
      for (int j = 0; j < 2; ++j)
        acc[i][j] = wmma16(afr[i].v, bfr[j].v, acc[i][j]);

    __syncthreads();
  }

  // ---- epilogue: C/D layout -> lane l16 = N, VGPR r -> M = r + 8*half ----
#pragma unroll
  for (int i = 0; i < 4; ++i) {
#pragma unroll
    for (int j = 0; j < 2; ++j) {
      const int row0 = m0 + waveM + i * 16 + half * 8;
      const int col  = n0 + waveN + j * 16 + l16;
      const float bv = bias ? bias[col] : 0.0f;
#pragma unroll
      for (int r = 0; r < 8; ++r)
        Cout[(size_t)(row0 + r) * N + col] = acc[i][j][r] + bv;
    }
  }
}

// ---------------------------------------------------------------------------
// RoPE + head split.  qkv fp32 [B,L, 3*C] -> q,k f16 [B,H,L,D] (rope applied,
// q pre-scaled by D^-1/2) and v f16 transposed [B,H,D,L] (so attention's P*V
// B-fragments read two runs of 8 contiguous key indices).  One thread per (b,l,h,d/2).
// ---------------------------------------------------------------------------
__global__ __launch_bounds__(256)
void rope_split(const float* __restrict__ qkv,
                _Float16* __restrict__ qh,
                _Float16* __restrict__ kh,
                _Float16* __restrict__ vt)
{
  const int idx = blockIdx.x * blockDim.x + threadIdx.x;
  const int d2 = idx & 31;
  const int h  = (idx >> 5) & (HH - 1);
  const int l  = (idx >> 9) & (LL - 1);
  const int b  = idx >> 20;

  const size_t rowbase = (size_t)(b * LL + l) * (3 * CC);
  const int    coloff  = h * DD + 2 * d2;

  const float qr = qkv[rowbase + 0 * CC + coloff];
  const float qi = qkv[rowbase + 0 * CC + coloff + 1];
  const float kr = qkv[rowbase + 1 * CC + coloff];
  const float ki = qkv[rowbase + 1 * CC + coloff + 1];
  const float vr = qkv[rowbase + 2 * CC + coloff];
  const float vi = qkv[rowbase + 2 * CC + coloff + 1];

  // theta_i = ROPE_SCALE / ROPE_BASE^(2i/D)
  const float theta = 3.0f * __powf(10000.0f, -(float)(2 * d2) / (float)DD);
  const float ang = (float)l * theta;
  float sn, cs;
  __sincosf(ang, &sn, &cs);

  const float qscale = 0.125f;  // D^-1/2, folded into q
  const float qor = (qr * cs - qi * sn) * qscale;
  const float qoi = (qr * sn + qi * cs) * qscale;
  const float kor = kr * cs - ki * sn;
  const float koi = kr * sn + ki * cs;

  const size_t qko = ((size_t)(b * HH + h) * LL + l) * DD + 2 * d2;
  qh[qko]     = (_Float16)qor;
  qh[qko + 1] = (_Float16)qoi;
  kh[qko]     = (_Float16)kor;
  kh[qko + 1] = (_Float16)koi;

  const size_t vto = ((size_t)(b * HH + h) * DD + 2 * d2) * LL + l;
  vt[vto]      = (_Float16)vr;
  vt[vto + LL] = (_Float16)vi;
}

// ---------------------------------------------------------------------------
// Flash attention.  Grid (L/64, B*H), 128 threads = 4 waves; each wave owns
// 16 query rows.  Per 32 keys: 4 WMMAs for S = Q*K^T (K-dim = D = 64 split
// in 2), online softmax (shfl_xor over 16-lane halves = C/D row striping),
// P relayout D->A through per-wave LDS, 4 WMMAs for O += P*V.
// Output written f16 in [B,L,C] layout ready for the projection GEMM.
// ---------------------------------------------------------------------------
__global__ __launch_bounds__(128)
void flash_attn(const _Float16* __restrict__ qh,
                const _Float16* __restrict__ kh,
                const _Float16* __restrict__ vt,
                _Float16* __restrict__ oh)
{
  const int bh   = blockIdx.y;            // b*H + h
  const int b    = bh >> 4;
  const int h    = bh & (HH - 1);
  const int wave = threadIdx.x >> 5;
  const int lane = threadIdx.x & 31;
  const int half = lane >> 4;
  const int l16  = lane & 15;
  const int q0   = blockIdx.x * 64 + wave * 16;

  __shared__ __align__(16) _Float16 Pl[4][16][40];   // per-wave P scratch

  const _Float16* qbase = qh + (size_t)bh * LL * DD;
  const _Float16* kbase = kh + (size_t)bh * LL * DD;
  const _Float16* vbase = vt + (size_t)bh * DD * LL;

  // Q A-fragments (d 0..31 and 32..63), resident for the whole loop.
  HFrag qa0, qa1;
  {
    const _Float16* qr = qbase + (size_t)(q0 + l16) * DD;
    const int kb = half * 8;
    qa0.u[0] = *(const uint4*)&qr[kb];
    qa0.u[1] = *(const uint4*)&qr[kb + 16];
    qa1.u[0] = *(const uint4*)&qr[32 + kb];
    qa1.u[1] = *(const uint4*)&qr[32 + kb + 16];
  }

  v8f acc[4];
#pragma unroll
  for (int c = 0; c < 4; ++c) acc[c] = v8f{};
  float m[8], ssum[8];
#pragma unroll
  for (int r = 0; r < 8; ++r) { m[r] = -1e30f; ssum[r] = 0.0f; }

  for (int j = 0; j < LL; j += 32) {
    // ---- scores S (16q x 32k), two 16x16 tiles ----
    v8f S[2];
#pragma unroll
    for (int t = 0; t < 2; ++t) {
      const _Float16* kr = kbase + (size_t)(j + t * 16 + l16) * DD;
      const int kb = half * 8;
      HFrag blo, bhi;                       // B[kk][n] = K[row n][d kk]
      blo.u[0] = *(const uint4*)&kr[kb];
      blo.u[1] = *(const uint4*)&kr[kb + 16];
      bhi.u[0] = *(const uint4*)&kr[32 + kb];
      bhi.u[1] = *(const uint4*)&kr[32 + kb + 16];
      v8f z = v8f{};
      z = wmma16(qa0.v, blo.v, z);
      S[t] = wmma16(qa1.v, bhi.v, z);
    }

    // ---- online softmax (rows live on 16-lane halves) ----
    float p0[8], p1[8], alpha[8];
#pragma unroll
    for (int r = 0; r < 8; ++r) {
      float mx = fmaxf(S[0][r], S[1][r]);
      mx = fmaxf(mx, __shfl_xor(mx, 1, 32));
      mx = fmaxf(mx, __shfl_xor(mx, 2, 32));
      mx = fmaxf(mx, __shfl_xor(mx, 4, 32));
      mx = fmaxf(mx, __shfl_xor(mx, 8, 32));
      const float mn = fmaxf(m[r], mx);
      alpha[r] = __expf(m[r] - mn);
      p0[r] = __expf(S[0][r] - mn);
      p1[r] = __expf(S[1][r] - mn);
      float ps = p0[r] + p1[r];
      ps += __shfl_xor(ps, 1, 32);
      ps += __shfl_xor(ps, 2, 32);
      ps += __shfl_xor(ps, 4, 32);
      ps += __shfl_xor(ps, 8, 32);
      ssum[r] = ssum[r] * alpha[r] + ps;
      m[r] = mn;
    }
#pragma unroll
    for (int c = 0; c < 4; ++c)
#pragma unroll
      for (int r = 0; r < 8; ++r) acc[c][r] *= alpha[r];

    // ---- relayout P (C/D layout) -> A layout via per-wave LDS ----
    // Same-wave DS ops are processed in order, so no barrier is needed.
#pragma unroll
    for (int r = 0; r < 8; ++r) {
      const int row = r + half * 8;
      Pl[wave][row][l16]      = (_Float16)p0[r];
      Pl[wave][row][16 + l16] = (_Float16)p1[r];
    }
    HFrag pa;
    {
      const int kb = half * 8;
      pa.u[0] = *(const uint4*)&Pl[wave][l16][kb];
      pa.u[1] = *(const uint4*)&Pl[wave][l16][kb + 16];
    }

    // ---- O += P * V ;  V^T layout gives contiguous key runs per lane ----
#pragma unroll
    for (int c = 0; c < 4; ++c) {
      const _Float16* vr = vbase + (size_t)(c * 16 + l16) * LL + j + half * 8;
      HFrag vb;                              // B[kk][n] = V[key kk][d c*16+n]
      vb.u[0] = *(const uint4*)&vr[0];
      vb.u[1] = *(const uint4*)&vr[16];
      acc[c] = wmma16(pa.v, vb.v, acc[c]);
    }
  }

  // ---- epilogue: normalize and emit f16 in [B,L,C] for the proj GEMM ----
#pragma unroll
  for (int c = 0; c < 4; ++c) {
#pragma unroll
    for (int r = 0; r < 8; ++r) {
      const int row = q0 + r + half * 8;
      const int col = h * DD + c * 16 + l16;
      oh[((size_t)b * LL + row) * CC + col] = (_Float16)(acc[c][r] / ssum[r]);
    }
  }
}

// ---------------------------------------------------------------------------
// Launch.  Workspace layout (~184 MB total):
//   qkv   f32 [8192][3072]   96 MB
//   qh    f16 [B,H,L,D]      16 MB
//   kh    f16 [B,H,L,D]      16 MB
//   vt    f16 [B,H,D,L]      16 MB
//   oh    f16 [B,L,C]        16 MB
//   xh    f16 [8192][1024]   16 MB
//   wqkvT f16 [3072][1024]    6 MB
//   wprjT f16 [1024][1024]    2 MB
// ---------------------------------------------------------------------------
extern "C" void kernel_launch(void* const* d_in, const int* in_sizes, int n_in,
                              void* d_out, int out_size, void* d_ws, size_t ws_size,
                              hipStream_t stream)
{
  const float* x      = (const float*)d_in[0];
  const float* w_qkv  = (const float*)d_in[1];
  const float* w_proj = (const float*)d_in[2];
  const float* b_proj = (const float*)d_in[3];
  float*       out    = (float*)d_out;

  char* ws = (char*)d_ws;
  float*    qkv   = (float*)ws;     ws += (size_t)BB * LL * 3 * CC * sizeof(float);
  _Float16* qh    = (_Float16*)ws;  ws += (size_t)BB * HH * LL * DD * sizeof(_Float16);
  _Float16* kh    = (_Float16*)ws;  ws += (size_t)BB * HH * LL * DD * sizeof(_Float16);
  _Float16* vt    = (_Float16*)ws;  ws += (size_t)BB * HH * DD * LL * sizeof(_Float16);
  _Float16* oh    = (_Float16*)ws;  ws += (size_t)BB * LL * CC * sizeof(_Float16);
  _Float16* xh    = (_Float16*)ws;  ws += (size_t)BB * LL * CC * sizeof(_Float16);
  _Float16* wqkvT = (_Float16*)ws;  ws += (size_t)CC * 3 * CC * sizeof(_Float16);
  _Float16* wprjT = (_Float16*)ws;

  const int M = BB * LL;        // 8192

  // 0) f16 conversions / weight transposes
  cvt_f16<<<(M * CC) / 1024, 256, 0, stream>>>(x, xh);
  {
    dim3 gq(3 * CC / 32, CC / 32);
    transpose_to_f16<<<gq, 256, 0, stream>>>(w_qkv, wqkvT, CC, 3 * CC);
    dim3 gp(CC / 32, CC / 32);
    transpose_to_f16<<<gp, 256, 0, stream>>>(w_proj, wprjT, CC, CC);
  }

  // 1) QKV GEMM: qkv = xh @ wqkvT^T   (M=8192, N=3072, K=1024)
  {
    dim3 grid(3 * CC / 128, M / 128);
    gemm_wmma_f16<<<grid, 256, 0, stream>>>(xh, wqkvT, qkv, nullptr,
                                            M, 3 * CC, CC);
  }

  // 2) RoPE + head split (+ fold D^-1/2 into q, transpose v)
  {
    const int total = BB * LL * HH * (DD / 2);   // 4,194,304
    rope_split<<<total / 256, 256, 0, stream>>>(qkv, qh, kh, vt);
  }

  // 3) Flash attention
  {
    dim3 grid(LL / 64, BB * HH);
    flash_attn<<<grid, 128, 0, stream>>>(qh, kh, vt, oh);
  }

  // 4) Projection GEMM + bias: out = oh @ wprjT^T + b_proj (M=8192, N=K=1024)
  {
    dim3 grid(CC / 128, M / 128);
    gemm_wmma_f16<<<grid, 256, 0, stream>>>(oh, wprjT, out, b_proj,
                                            M, CC, CC);
  }
}
